// Correlation_65652870087538
// MI455X (gfx1250) — matrix operand order repaired and checked by reading, not compiled, not run on hardware
//
#include <hip/hip_runtime.h>
#include <hip/hip_bf16.h>
#include <stdint.h>

// ---------------------------------------------------------------------------
// FlowNetC correlation on gfx1250 (MI455X), v_wmma_f32_16x16x32_bf16 +
// global_load_async_to_lds_b128 double-buffered staging + LDS-staged
// coalesced output extraction.
//
//  out[b, dy*21+dx, h, w] = (1/256) * sum_c x1[b,c,h,w] * x2p[b,c,h+2dy,w+2dx]
//
// Fast path (needs 44 MB workspace):
//   1) convert kernels: f32 NCHW -> bf16 channel-last; x2 pre-padded to
//      width 112 (xpad = x+20, zeros outside) so staging needs no predication.
//   2) main kernel: per (b,h,16-px tile): A-fragments (X1) resident in VGPRs;
//      per dy, async-DMA one padded X2 row (64 cols x 256 ch bf16, contiguous
//      512B rows) into LDS (double buffered, overlapped with compute);
//      P[m,j] = sum_c X1[c,m]*X2[c,j] via 8 software-pipelined WMMAs per
//      16x16 N-tile (one tile per wave); dump P to LDS and store the
//      diagonals out[dx,m] = P[m, m+2dx] as coalesced float4 runs.
// ---------------------------------------------------------------------------

typedef __attribute__((ext_vector_type(16))) __bf16 bf16x16;
typedef __attribute__((ext_vector_type(8)))  float  f32x8;

#define C_CH   256
#define HW     64
#define NDY    21
#define NDX    21
#define TILE_W 16
#define XPADW  112                      // padded x2 width: xpad in [0,112)
#define LDS_STRIDE 264                  // ushorts per LDS row (256 + 8 pad)
#define P_STRIDE   20                   // floats per sP row (16 + 4 pad)

#define X1BF_ELEMS ((size_t)8 * HW * HW * C_CH)            // [b][h][w][c]
#define X2BF_ELEMS ((size_t)8 * HW * XPADW * C_CH)         // [b][y][xpad][c]
#define WS_NEEDED  ((X1BF_ELEMS + X2BF_ELEMS) * sizeof(unsigned short))

union Frag {
  uint4   q[2];   // 32 bytes
  bf16x16 v;      // 16 x bf16
};

__device__ __forceinline__ unsigned short f2bf(float f) {
  unsigned u = __float_as_uint(f);
  unsigned r = u + 0x7FFFu + ((u >> 16) & 1u);   // round-to-nearest-even
  return (unsigned short)(r >> 16);
}

// Low 32 bits of a generic pointer to LDS = LDS byte address (flat aperture rule).
__device__ __forceinline__ uint32_t lds_off(const void* p) {
  return (uint32_t)(uintptr_t)p;
}

// CDNA5 async DMA: 16B gather from global into LDS, tracked by ASYNCcnt.
__device__ __forceinline__ void async_copy_b128(uint32_t lds_byte_addr,
                                                const void* gaddr) {
  asm volatile("global_load_async_to_lds_b128 %0, %1, off"
               :: "v"(lds_byte_addr), "v"(gaddr)
               : "memory");
}

__device__ __forceinline__ void wait_async0() {
  asm volatile("s_wait_asynccnt 0x0" ::: "memory");
}

// ------------------------- conversion pre-pass -----------------------------
// x1: f32 [b,c,h,w] -> bf16 [b,h,w,c]  (LDS-tiled transpose, coalesced both sides)
__global__ __launch_bounds__(256, 1)
void cvt_x1_kernel(const float* __restrict__ in1, unsigned short* __restrict__ o) {
  __shared__ unsigned short t[64 * 72];
  const int b = blockIdx.z, h = blockIdx.y, c0 = blockIdx.x * 64;
  {
    const int x = threadIdx.x & 63, cs = threadIdx.x >> 6;
    for (int cc = cs; cc < 64; cc += 4) {
      float v = in1[(((size_t)b * C_CH + c0 + cc) * HW + h) * HW + x];
      t[x * 72 + cc] = f2bf(v);
    }
  }
  __syncthreads();
  {
    const int cc = threadIdx.x & 63, wsub = threadIdx.x >> 6;
    for (int w = wsub; w < 64; w += 4)
      o[(((size_t)b * HW + h) * HW + w) * C_CH + c0 + cc] = t[w * 72 + cc];
  }
}

// x2: f32 [b,c,y,x] -> bf16 [b,y,xpad,c], xpad = x+20, zero borders.
__global__ __launch_bounds__(256, 1)
void cvt_x2_kernel(const float* __restrict__ in2, unsigned short* __restrict__ o) {
  __shared__ unsigned short t[64 * 72];
  const int b = blockIdx.z, y = blockIdx.y, c0 = blockIdx.x * 64;
  {
    const int x = threadIdx.x & 63, cs = threadIdx.x >> 6;
    for (int cc = cs; cc < 64; cc += 4) {
      float v = in2[(((size_t)b * C_CH + c0 + cc) * HW + y) * HW + x];
      t[x * 72 + cc] = f2bf(v);
    }
  }
  __syncthreads();
  {
    const int cc = threadIdx.x & 63, wsub = threadIdx.x >> 6;
    for (int xp = wsub; xp < XPADW; xp += 4) {
      const int x = xp - 20;
      unsigned short v = ((unsigned)x < (unsigned)HW) ? t[x * 72 + cc]
                                                      : (unsigned short)0;
      o[(((size_t)b * HW + y) * XPADW + xp) * C_CH + c0 + cc] = v;
    }
  }
}

// ------------------------------ main kernel --------------------------------
__global__ __launch_bounds__(128, 1)
void corr_wmma_bf16_kernel(const unsigned short* __restrict__ x1bf,
                           const unsigned short* __restrict__ x2bf,
                           float* __restrict__ out) {
  __shared__ unsigned short sX1[16 * LDS_STRIDE];        //  8.4 KB  [m][c]
  __shared__ unsigned short sX2[2][64 * LDS_STRIDE];     // 67.6 KB  [j][c] x2
  __shared__ float          sP[64 * P_STRIDE];           //  5.1 KB  [j][m]

  const int tid  = threadIdx.x;
  const int lane = tid & 31;
  const int wave = tid >> 5;               // 0..3 -> N-tile of P
  const int w0   = blockIdx.x * TILE_W;
  const int h    = blockIdx.y;
  const int b    = blockIdx.z;

  // ---- stage X1 tile via async DMA: 16 rows x 4 x 16B (128B apart) = 64 x B128 ----
  if (tid < 64) {
    const int m = tid >> 2, ch = tid & 3;                // ch: 128B chunk
    const unsigned short* g =
        x1bf + (((size_t)b * HW + h) * HW + (w0 + m)) * C_CH + ch * 64;
    async_copy_b128(lds_off(&sX1[m * LDS_STRIDE + ch * 64]), g);
  }
  // ---- kick off dy=0 X2 row while X1 settles ----
  {
    const int y0 = h - 20;
    if ((unsigned)y0 < (unsigned)HW) {
      #pragma unroll
      for (int r = 0; r < 2; ++r) {
        const int id = tid + r * 128;                    // 256 chunks
        const int j = id >> 2, ch = id & 3;
        const unsigned short* g =
            x2bf + (((size_t)b * HW + y0) * XPADW + (w0 + j)) * C_CH + ch * 64;
        async_copy_b128(lds_off(&sX2[0][j * LDS_STRIDE + ch * 64]), g);
      }
    } else {
      const uint4 z = {0u, 0u, 0u, 0u};
      for (int k = tid; k < 64 * 32; k += 128)           // 64 rows x 32 uint4
        *(uint4*)&sX2[0][(k >> 5) * LDS_STRIDE + (k & 31) * 8] = z;
    }
  }
  wait_async0();
  __syncthreads();

  // ---- preload the 8 A fragments (K = 256 as 8 x 32) ----
  // A 16x32 bf16: lanes 0-15 -> M=lane, K 0..7/16..23; lanes 16-31 -> K 8..15/24..31
  Frag fa[8];
  {
    const int m = lane & 15, sel = lane >> 4;
    #pragma unroll
    for (int ks = 0; ks < 8; ++ks) {
      const unsigned short* plo = &sX1[m * LDS_STRIDE + ks * 32 + sel * 8];
      fa[ks].q[0] = *(const uint4*)plo;
      fa[ks].q[1] = *(const uint4*)(plo + 16);
    }
  }

  // B 32x16 bf16: lanes 0-15 -> N=lane, K 0..7/16..23; lanes 16-31 -> K 8..15/24..31
  const int jt    = wave * 16;
  const int n     = lane & 15;
  const int selB  = lane >> 4;
  const int mbase = selB * 8;              // D: VGPR r -> M = r + 8*(lane>=16)
  const unsigned short* pbBase0 = &sX2[0][(jt + n) * LDS_STRIDE + selB * 8];
  const unsigned short* pbBase1 = &sX2[1][(jt + n) * LDS_STRIDE + selB * 8];

  for (int dy = 0; dy < NDY; ++dy) {
    const unsigned short* pbBase = (dy & 1) ? pbBase1 : pbBase0;

    // issue next row's async copy into the other buffer (overlaps compute)
    if (dy + 1 < NDY) {
      const int slot = (dy + 1) & 1;
      const int y = h + 2 * (dy + 1) - 20;
      if ((unsigned)y < (unsigned)HW) {
        #pragma unroll
        for (int r = 0; r < 2; ++r) {
          const int id = tid + r * 128;
          const int j = id >> 2, ch = id & 3;
          const unsigned short* g =
              x2bf + (((size_t)b * HW + y) * XPADW + (w0 + j)) * C_CH + ch * 64;
          async_copy_b128(lds_off(&sX2[slot][j * LDS_STRIDE + ch * 64]), g);
        }
      } else {
        const uint4 z = {0u, 0u, 0u, 0u};
        for (int k = tid; k < 64 * 32; k += 128)
          *(uint4*)&sX2[slot][(k >> 5) * LDS_STRIDE + (k & 31) * 8] = z;
      }
    }

    // ---- software-pipelined B loads + 8 WMMAs ----
    f32x8 acc = {0.f, 0.f, 0.f, 0.f, 0.f, 0.f, 0.f, 0.f};
    Frag cur, nxt;
    cur.q[0] = *(const uint4*)(pbBase);
    cur.q[1] = *(const uint4*)(pbBase + 16);
    #pragma unroll
    for (int ks = 0; ks < 8; ++ks) {
      if (ks + 1 < 8) {
        const unsigned short* pb = pbBase + (ks + 1) * 32;
        nxt.q[0] = *(const uint4*)pb;
        nxt.q[1] = *(const uint4*)(pb + 16);
      }
      acc = __builtin_amdgcn_wmma_f32_16x16x32_bf16(
          false, fa[ks].v, false, cur.v, (short)0, acc, false, false);
      if (ks + 1 < 8) cur = nxt;
    }

    // ---- dump P tile to LDS (conflict-padded), then coalesced store ----
    {
      float4 lo, hi;
      lo.x = acc[0]; lo.y = acc[1]; lo.z = acc[2]; lo.w = acc[3];
      hi.x = acc[4]; hi.y = acc[5]; hi.z = acc[6]; hi.w = acc[7];
      float* row = &sP[(jt + n) * P_STRIDE + mbase];
      *(float4*)(row)     = lo;     // P[mbase..+3 , jt+n]
      *(float4*)(row + 4) = hi;     // P[mbase+4..+7, jt+n]
    }
    __syncthreads();

    if (tid < 84) {                         // 21 dx * 4 groups of 4 m
      const int dx = tid >> 2, mg = (tid & 3) * 4;
      float4 v;
      v.x = sP[(mg + 0 + 2 * dx) * P_STRIDE + mg + 0] * (1.0f / 256.0f);
      v.y = sP[(mg + 1 + 2 * dx) * P_STRIDE + mg + 1] * (1.0f / 256.0f);
      v.z = sP[(mg + 2 + 2 * dx) * P_STRIDE + mg + 2] * (1.0f / 256.0f);
      v.w = sP[(mg + 3 + 2 * dx) * P_STRIDE + mg + 3] * (1.0f / 256.0f);
      const int d = dy * NDX + dx;
      *(float4*)&out[(((size_t)b * (NDY * NDX) + d) * HW + h) * HW + w0 + mg] = v;
    }

    wait_async0();       // next buffer's DMA complete
    __syncthreads();     // everyone done reading current buffer + sP
  }
}

// --------------------- fallback (no-workspace) kernel ----------------------
__global__ __launch_bounds__(128, 1)
void corr_fallback_kernel(const float* __restrict__ in1,
                          const float* __restrict__ in2,
                          float* __restrict__ out) {
  __shared__ unsigned short sX1[16 * LDS_STRIDE];
  __shared__ unsigned short sX2[64 * LDS_STRIDE];

  const int tid = threadIdx.x, lane = tid & 31, wave = tid >> 5;
  const int w0 = blockIdx.x * TILE_W, h = blockIdx.y, b = blockIdx.z;
  const size_t planeB = (size_t)b * C_CH * HW * HW;

  {
    const int m = tid & 15;
    for (int c = tid >> 4; c < C_CH; c += 8)
      sX1[m * LDS_STRIDE + c] =
          f2bf(in1[planeB + (size_t)c * (HW * HW) + (size_t)h * HW + (w0 + m)]);
  }
  __syncthreads();

  Frag fa[8];
  {
    const int m = lane & 15, sel = lane >> 4;
    #pragma unroll
    for (int ks = 0; ks < 8; ++ks) {
      const unsigned short* plo = &sX1[m * LDS_STRIDE + ks * 32 + sel * 8];
      fa[ks].q[0] = *(const uint4*)plo;
      fa[ks].q[1] = *(const uint4*)(plo + 16);
    }
  }

  const int jt = wave * 16, n = lane & 15, selB = lane >> 4, mbase = (lane >> 4) * 8;

  for (int dy = 0; dy < NDY; ++dy) {
    __syncthreads();
    {
      const int j = tid & 63, x = w0 + j - 20, y = h + 2 * dy - 20;
      const bool ok = ((unsigned)x < (unsigned)HW) && ((unsigned)y < (unsigned)HW);
      const float* src = in2 + planeB + (size_t)y * HW + x;
      for (int c = tid >> 6; c < C_CH; c += 2)
        sX2[j * LDS_STRIDE + c] = f2bf(ok ? src[(size_t)c * (HW * HW)] : 0.0f);
    }
    __syncthreads();

    f32x8 acc = {0.f, 0.f, 0.f, 0.f, 0.f, 0.f, 0.f, 0.f};
    #pragma unroll
    for (int ks = 0; ks < 8; ++ks) {
      Frag fb;
      const unsigned short* pb = &sX2[(jt + n) * LDS_STRIDE + ks * 32 + selB * 8];
      fb.q[0] = *(const uint4*)pb;
      fb.q[1] = *(const uint4*)(pb + 16);
      acc = __builtin_amdgcn_wmma_f32_16x16x32_bf16(
          false, fa[ks].v, false, fb.v, (short)0, acc, false, false);
    }
    #pragma unroll
    for (int r = 0; r < 8; ++r) {
      const int m = mbase + r, dj = (jt + n) - m;
      if (dj >= 0 && dj <= 40 && ((dj & 1) == 0)) {
        const int d = dy * NDX + (dj >> 1);
        out[(((size_t)b * (NDY * NDX) + d) * HW + h) * HW + (w0 + m)] =
            acc[r] * (1.0f / 256.0f);
      }
    }
  }
}

// ------------------------------- launcher ----------------------------------
extern "C" void kernel_launch(void* const* d_in, const int* in_sizes, int n_in,
                              void* d_out, int out_size, void* d_ws, size_t ws_size,
                              hipStream_t stream) {
  const float* in1 = (const float*)d_in[0];
  const float* in2 = (const float*)d_in[1];
  float* out = (float*)d_out;
  (void)in_sizes; (void)n_in; (void)out_size;

  dim3 grid(HW / TILE_W, HW, 8);     // (4, 64, 8)
  dim3 block(128);

  if (d_ws != nullptr && ws_size >= WS_NEEDED) {
    unsigned short* x1bf = (unsigned short*)d_ws;
    unsigned short* x2bf = x1bf + X1BF_ELEMS;
    dim3 cgrid(C_CH / 64, HW, 8);    // (4, 64, 8)
    dim3 cblk(256);
    cvt_x1_kernel<<<cgrid, cblk, 0, stream>>>(in1, x1bf);
    cvt_x2_kernel<<<cgrid, cblk, 0, stream>>>(in2, x2bf);
    corr_wmma_bf16_kernel<<<grid, block, 0, stream>>>(x1bf, x2bf, out);
  } else {
    corr_fallback_kernel<<<grid, block, 0, stream>>>(in1, in2, out);
  }
}
